// SSMRTF_model_10557029614110
// MI455X (gfx1250) — compile-verified, matrix operand order checked by reading, not run on hardware
//
#include <hip/hip_runtime.h>

// ---------------------------------------------------------------------------
// SSM-RTF forward:  y = relu( causal_conv(x, taps(A,C)) )
// Lower-triangular Toeplitz GEMM on CDNA5 bf16 WMMA, with CDNA5 async
// global->LDS DMA (GLOBAL_LOAD_ASYNC_TO_LDS_B128, ASYNCcnt) for staging:
// no VGPR tile staging => no spills around the 128-VGPR accumulator block.
// ---------------------------------------------------------------------------

typedef __bf16 bf16;
typedef bf16  v16bf __attribute__((ext_vector_type(16)));
typedef float v8f   __attribute__((ext_vector_type(8)));

#define SEQ_L 4096
#define DCH   1024
#define NBATCH 4
#define NP1   65            // HIDDEN + 1

#define BM 128
#define BN 128
#define BK 32
#define APITCH 40           // lds pitch (bf16): 80 B, 16B-aligned, conflict-free
#define CHUNK_STRIDE 2560   // 32 rows * 80 B between staging chunk groups
#define TAPS_PAD 128
#define NTILES 128          // doff = (tile_m - kb)/32 in [-3, 124] -> 128 tiles
#define TILE_ELEMS (BM * BK)

union BF2 { unsigned u; bf16 h[2]; };

// generic LDS pointer -> 32-bit LDS byte address (low 32 bits of the generic address)
static __device__ __forceinline__ unsigned lds_off(const void* p) {
    return (unsigned)(unsigned long long)p;
}

#define WAIT_ASYNC_8() asm volatile("s_wait_asynccnt 0x8" ::: "memory")
#define WAIT_ASYNC_0() asm volatile("s_wait_asynccnt 0x0" ::: "memory")

// Issue one tile (A: 4 chunks, B: 4 chunks per thread) as async LDS DMA.
static __device__ __forceinline__ void issue_tile_async(
    const char* gA, const char* gB, unsigned la, unsigned lb, int tid)
{
#pragma unroll
    for (int c = 0; c < 4; ++c) {
        asm volatile("global_load_async_to_lds_b128 %0, %1, off"
                     :: "v"(la + (unsigned)(c * CHUNK_STRIDE)),
                        "v"((unsigned long long)(gA + (tid + 128 * c) * 16))
                     : "memory");
        asm volatile("global_load_async_to_lds_b128 %0, %1, off"
                     :: "v"(lb + (unsigned)(c * CHUNK_STRIDE)),
                        "v"((unsigned long long)(gB + (tid + 128 * c) * 16))
                     : "memory");
    }
}

// ---------------------------------------------------------------------------
// Kernel 1: taps recurrence + zero pads (sequential; one small block)
// ---------------------------------------------------------------------------
__global__ __launch_bounds__(96)
void rtf_taps_kernel(const float* __restrict__ A,
                     const float* __restrict__ C,
                     float* __restrict__ tapspad)
{
    __shared__ float num[NP1];
    __shared__ float av[NP1];
    const int tid = threadIdx.x;
    if (tid < NP1) { num[tid] = C[tid]; av[tid] = A[tid]; }
    for (int i = tid; i < TAPS_PAD; i += 96) {
        tapspad[i] = 0.0f;                          // front pad
        tapspad[TAPS_PAD + SEQ_L + i] = 0.0f;       // back pad
    }
    __syncthreads();
    for (int t = 0; t < SEQ_L; ++t) {
        const float k0 = num[0];
        float tmp = 0.0f;
        if (tid < NP1) tmp = num[tid] - av[tid] * k0;
        __syncthreads();
        if (tid == 0) { tapspad[TAPS_PAD + t] = k0; num[NP1 - 1] = tmp; }
        else if (tid < NP1) { num[tid - 1] = tmp; }
        __syncthreads();
    }
}

// ---------------------------------------------------------------------------
// Kernel 2: materialize the 128 distinct Toeplitz tiles in bf16, k-major.
// ---------------------------------------------------------------------------
__global__ __launch_bounds__(256)
void build_atiles_kernel(const float* __restrict__ tapspad,
                         bf16* __restrict__ atiles)
{
    const int doffIdx = blockIdx.x;                 // 0..127
    const int base = (doffIdx - 3) * BK;
    bf16* dst = atiles + (size_t)doffIdx * TILE_ELEMS;
#pragma unroll
    for (int r = 0; r < 16; ++r) {
        const int idx = threadIdx.x + 256 * r;      // 4096 elems
        const int i = idx >> 5;
        const int j = idx & 31;
        dst[idx] = (bf16)tapspad[TAPS_PAD + base + i - j];
    }
}

// ---------------------------------------------------------------------------
// Kernel 3: convert x fp32 -> bf16, retiled so every GEMM B-tile is one
// contiguous block:  xt[((b*128 + t/32)*1024 + d)*32 + t%32]
// ---------------------------------------------------------------------------
__global__ __launch_bounds__(256)
void convert_retile_x_kernel(const float* __restrict__ x,
                             bf16* __restrict__ xt)
{
    const int gid  = blockIdx.x * 256 + threadIdx.x;  // 4*2048*1024 threads
    const int d    = gid & (DCH - 1);
    const int rest = gid >> 10;
    const int p    = rest & (SEQ_L / 2 - 1);
    const int b    = rest >> 11;
    const int t0   = p * 2;
    const float f0 = x[((size_t)b * SEQ_L + t0) * DCH + d];
    const float f1 = x[((size_t)b * SEQ_L + t0 + 1) * DCH + d];
    BF2 pk; pk.h[0] = (bf16)f0; pk.h[1] = (bf16)f1;
    unsigned* dst = (unsigned*)(xt +
        ((((size_t)b * (SEQ_L / BK) + (t0 >> 5)) * DCH + d) * BK + (t0 & 31)));
    *dst = pk.u;
}

// ---------------------------------------------------------------------------
// One K-step of WMMA compute from a given LDS buffer pair.
// ---------------------------------------------------------------------------
static __device__ __forceinline__ void compute_step(
    const bf16 (*la)[APITCH], const bf16 (*lb)[APITCH],
    int wm, int wn, int row16, int koff, v8f (&acc)[4][4])
{
    union Frag { v16bf v; uint4 q[2]; };
    Frag afr[4], bfr[4];
#pragma unroll
    for (int mi = 0; mi < 4; ++mi) {
        const bf16* p = &la[wm * 64 + mi * 16 + row16][koff];
        afr[mi].q[0] = *(const uint4*)(p);
        afr[mi].q[1] = *(const uint4*)(p + 16);
    }
#pragma unroll
    for (int ni = 0; ni < 4; ++ni) {
        const bf16* p = &lb[wn * 64 + ni * 16 + row16][koff];
        bfr[ni].q[0] = *(const uint4*)(p);
        bfr[ni].q[1] = *(const uint4*)(p + 16);
    }
#pragma unroll
    for (int mi = 0; mi < 4; ++mi)
#pragma unroll
        for (int ni = 0; ni < 4; ++ni)
            acc[mi][ni] = __builtin_amdgcn_wmma_f32_16x16x32_bf16(
                false, afr[mi].v, false, bfr[ni].v,
                (short)0, acc[mi][ni], false, false);
}

// ---------------------------------------------------------------------------
// Kernel 4: Toeplitz GEMM + fused ReLU.
// 128 thr = 4 waves (2x2); block tile 128x128; wave tile 64x64 = 4x4 frags.
// Double-buffered LDS filled by async global->LDS DMA; split-counter wait.
// ---------------------------------------------------------------------------
__global__ __launch_bounds__(128)
void toeplitz_gemm_relu_kernel(const bf16* __restrict__ atiles,
                               const bf16* __restrict__ xt,
                               float* __restrict__ y)
{
    __shared__ alignas(16) bf16 lds_a[2][BM][APITCH];
    __shared__ alignas(16) bf16 lds_b[2][BN][APITCH];

    const int tid  = threadIdx.x;
    const int lane = tid & 31;
    const int wave = tid >> 5;
    const int wm   = wave >> 1;            // 0..1 -> 64-row slab
    const int wn   = wave & 1;             // 0..1 -> 64-col slab

    const int tile_n = blockIdx.x * BN;
    const int tile_m = blockIdx.y * BM;
    const int bidx   = tile_n >> 10;
    const int d0     = tile_n & (DCH - 1);

    const int row16 = lane & 15;
    const int koff  = (lane & 16) ? 8 : 0;
    const int mrow  = (lane & 16) ? 8 : 0;

    // per-thread LDS staging base addresses (byte), per buffer
    const unsigned la0 = lds_off(&lds_a[0][tid >> 2][(tid & 3) * 8]);
    const unsigned la1 = lds_off(&lds_a[1][tid >> 2][(tid & 3) * 8]);
    const unsigned lb0 = lds_off(&lds_b[0][tid >> 2][(tid & 3) * 8]);
    const unsigned lb1 = lds_off(&lds_b[1][tid >> 2][(tid & 3) * 8]);

    v8f acc[4][4];
#pragma unroll
    for (int mi = 0; mi < 4; ++mi)
#pragma unroll
        for (int ni = 0; ni < 4; ++ni)
            acc[mi][ni] = (v8f){0.f,0.f,0.f,0.f,0.f,0.f,0.f,0.f};

    // k_end = tile_m + 128 <= 4096 always; K-step count is even.
    const int k_end = tile_m + BM;

    // global byte addresses for tile at kb
    const char* aBase = (const char*)atiles;
    const char* bBase = (const char*)(xt + ((size_t)bidx * (SEQ_L / BK) * DCH + d0) * BK);
#define GA(kb) (aBase + (size_t)((((tile_m - (kb)) >> 5) + 3)) * (TILE_ELEMS * 2))
#define GB(kb) (bBase + (size_t)((kb) >> 5) * (DCH * BK * 2))

    issue_tile_async(GA(0), GB(0), la0, lb0, tid);          // prologue -> buf0

    for (int kb = 0; kb < k_end; kb += 2 * BK) {
        // ---- phase 0: prefetch kb+32 -> buf1, compute buf0 ----
        issue_tile_async(GA(kb + BK), GB(kb + BK), la1, lb1, tid);
        WAIT_ASYNC_8();                 // own wave's buf0 fill complete
        __syncthreads();                // everyone's buf0 fill visible
        compute_step(lds_a[0], lds_b[0], wm, wn, row16, koff, acc);
        __syncthreads();                // all readers done with buf0

        // ---- phase 1: prefetch kb+64 -> buf0 (if any), compute buf1 ----
        const int kn2 = kb + 2 * BK;
        if (kn2 < k_end) {
            issue_tile_async(GA(kn2), GB(kn2), la0, lb0, tid);
            WAIT_ASYNC_8();
        } else {
            WAIT_ASYNC_0();
        }
        __syncthreads();
        compute_step(lds_a[1], lds_b[1], wm, wn, row16, koff, acc);
        __syncthreads();
    }
#undef GA
#undef GB

    // epilogue: fused ReLU, fp32 store
#pragma unroll
    for (int mi = 0; mi < 4; ++mi) {
#pragma unroll
        for (int ni = 0; ni < 4; ++ni) {
#pragma unroll
            for (int r = 0; r < 8; ++r) {
                const int m = tile_m + wm * 64 + mi * 16 + mrow + r;
                const int d = d0 + wn * 64 + ni * 16 + row16;
                y[((size_t)bidx * SEQ_L + m) * DCH + d] =
                    fmaxf(acc[mi][ni][r], 0.0f);
            }
        }
    }
}

// ---------------------------------------------------------------------------
extern "C" void kernel_launch(void* const* d_in, const int* in_sizes, int n_in,
                              void* d_out, int out_size, void* d_ws, size_t ws_size,
                              hipStream_t stream)
{
    const float* x = (const float*)d_in[0];   // (4, 4096, 1024) fp32
    const float* A = (const float*)d_in[1];   // (1, 65) fp32
    const float* C = (const float*)d_in[2];   // (1, 65) fp32
    float* y = (float*)d_out;                 // (4, 4096, 1024) fp32

    // workspace layout
    float* tapspad = (float*)d_ws;                              // 4352 fp32
    bf16*  atiles  = (bf16*)((char*)d_ws + 20480);              // 1 MB
    bf16*  xt      = (bf16*)((char*)d_ws + 20480 + (size_t)NTILES * TILE_ELEMS * 2); // 32 MB

    rtf_taps_kernel<<<1, 96, 0, stream>>>(A, C, tapspad);
    build_atiles_kernel<<<NTILES, 256, 0, stream>>>(tapspad, atiles);
    convert_retile_x_kernel<<<(NBATCH * (SEQ_L / 2) * DCH) / 256, 256, 0, stream>>>(x, xt);

    dim3 grid((NBATCH * DCH) / BN, SEQ_L / BM);   // (32, 32)
    toeplitz_gemm_relu_kernel<<<grid, 128, 0, stream>>>(atiles, xt, y);
}
